// VQLayer_33457795235998
// MI455X (gfx1250) — matrix operand order repaired and checked by reading, not compiled, not run on hardware
//
#include <hip/hip_runtime.h>
#include <hip/hip_bf16.h>
#include <stdint.h>

// Problem sizes (fixed by the reference)
#define DD     512
#define NROWS  16384      // 16 * 1024
#define VOCAB  8192
#define CHUNK  128        // codebook rows per chunk
#define KHALF  256        // K elements staged per phase (half of DD)
#define BUFBYTES 131072   // one phase buffer: hi 64KB + lo 64KB

typedef __attribute__((ext_vector_type(16))) __bf16 v16bf;
typedef __attribute__((ext_vector_type(8)))  __bf16 v8bf;
typedef __attribute__((ext_vector_type(4)))  __bf16 v4bf;
typedef __attribute__((ext_vector_type(8)))  float  v8f;

union V16U { v16bf v; v8bf h[2]; };

static __device__ __forceinline__ v8f wmma_bf16(v16bf a, v16bf b, v8f c) {
  return __builtin_amdgcn_wmma_f32_16x16x32_bf16(false, a, false, b, (short)0, c,
                                                 false, false);
}

// Async global -> LDS, 16 B per lane per instruction (tracked by ASYNCcnt).
static __device__ __forceinline__ void async_b128(uint32_t lds_off, const void* g) {
  asm volatile("global_load_async_to_lds_b128 %0, %1, off"
               :: "v"(lds_off), "v"(g) : "memory");
}
static __device__ __forceinline__ void wait_async0() {
  asm volatile("s_wait_asynccnt 0" ::: "memory");
}

// ---------------------------------------------------------------------------
// Prep: X (fp32) -> xh/xl (bf16 hi/lo split). One float4 per thread.
// ---------------------------------------------------------------------------
__global__ __launch_bounds__(256) void vq_prep_x(const float* __restrict__ x,
                                                 __bf16* __restrict__ xh,
                                                 __bf16* __restrict__ xl) {
  int gid = blockIdx.x * 256 + threadIdx.x;          // one float4
  float4 f = ((const float4*)x)[gid];
  float fs[4] = {f.x, f.y, f.z, f.w};
  v4bf h, l;
#pragma unroll
  for (int t = 0; t < 4; ++t) {
    __bf16 hv = (__bf16)fs[t];
    h[t] = hv;
    l[t] = (__bf16)(fs[t] - (float)hv);
  }
  ((v4bf*)xh)[gid] = h;
  ((v4bf*)xl)[gid] = l;
}

// ---------------------------------------------------------------------------
// Prep: codebook -> ch/cl (bf16 hi/lo) + c2 = ||c||^2, zero used[] flags.
// One wave per codebook row.
// ---------------------------------------------------------------------------
__global__ __launch_bounds__(256) void vq_prep_cb(const float* __restrict__ cb,
                                                  __bf16* __restrict__ ch,
                                                  __bf16* __restrict__ cl,
                                                  float* __restrict__ c2,
                                                  unsigned* __restrict__ used) {
  const int wave = threadIdx.x >> 5, lane = threadIdx.x & 31;
  const int row  = blockIdx.x * 8 + wave;
  const float4* src = (const float4*)(cb + (size_t)row * DD);
  v4bf* dh = (v4bf*)(ch + (size_t)row * DD);
  v4bf* dl = (v4bf*)(cl + (size_t)row * DD);
  float s = 0.f;
#pragma unroll
  for (int j = lane; j < DD / 4; j += 32) {
    float4 f = src[j];
    float fs[4] = {f.x, f.y, f.z, f.w};
    v4bf h, l;
#pragma unroll
    for (int t = 0; t < 4; ++t) {
      __bf16 hv = (__bf16)fs[t];
      h[t] = hv;
      l[t] = (__bf16)(fs[t] - (float)hv);
      s += fs[t] * fs[t];
    }
    dh[j] = h;
    dl[j] = l;
  }
#pragma unroll
  for (int off = 16; off; off >>= 1) s += __shfl_xor(s, off, 32);
  if (lane == 0) {
    c2[row]   = s;
    used[row] = 0u;
  }
}

// ---------------------------------------------------------------------------
// Main: fused (||c||^2 - 2 X.C^T) argmin over all codebook rows + gather.
// Grid: 128 blocks x 256 threads (8 waves = 4(M) x 2(N)); block owns 128 rows.
// Codebook streamed in 128-col chunks, each chunk staged as two K=256 phases,
// double-buffered in LDS via GLOBAL_LOAD_ASYNC_TO_LDS_B128 (overlaps WMMA).
// LDS: 2 x 128KB phase buffers + idx buffer.
//   buffer b: hi[128][256] bf16 at b*131072, lo[128][256] at b*131072+65536.
// ---------------------------------------------------------------------------
__global__ __launch_bounds__(256, 1) void vq_main(
    const __bf16* __restrict__ xh, const __bf16* __restrict__ xl,
    const __bf16* __restrict__ ch, const __bf16* __restrict__ cl,
    const float*  __restrict__ c2, const float*  __restrict__ cb,
    float* __restrict__ outEmbed, float* __restrict__ outIdx,
    unsigned* __restrict__ used) {
  extern __shared__ char smem[];
  int* idxS = (int*)(smem + 2 * BUFBYTES);

  const int tid   = threadIdx.x;
  const int lane  = tid & 31;
  const int wave  = tid >> 5;
  const int mwave = wave >> 1;                  // 0..3
  const int nwave = wave & 1;                   // 0..1
  const int half  = lane >> 4;                  // 0/1 (lane half)
  const int l15   = lane & 15;
  const int wgRow0 = blockIdx.x * 128;
  const int m0     = wgRow0 + mwave * 32 + l15; // mt==1 adds 16 rows

  // Issue async staging of chunk c0, K-half kh into buffer at bufbase bytes.
  // 4096 16B-units per array; 16 hi + 16 lo instructions per thread.
  auto stage = [&](int c0, int kh, uint32_t bufbase) {
#pragma unroll
    for (int i = 0; i < 16; ++i) {
      int e   = tid + i * 256;                  // 0..4095
      int row = e >> 5;                         // 0..127
      int un  = e & 31;                         // 16B unit within row
      const __bf16* gh = ch + (size_t)(c0 + row) * DD + kh * KHALF + un * 8;
      const __bf16* gl = cl + (size_t)(c0 + row) * DD + kh * KHALF + un * 8;
      uint32_t lo = bufbase + (uint32_t)(row * KHALF + un * 8) * 2;
      async_b128(lo,          gh);
      async_b128(lo + 65536,  gl);
    }
  };

  float best[2][8];
  int   bidx[2][8];
#pragma unroll
  for (int mt = 0; mt < 2; ++mt)
#pragma unroll
    for (int r = 0; r < 8; ++r) { best[mt][r] = 3.4e38f; bidx[mt][r] = 0; }

  // Prologue: stage phase 0 (chunk 0, K-half 0) into buffer 0.
  stage(0, 0, 0);
  wait_async0();
  __syncthreads();

  uint32_t pb = 0;   // current buffer base (bytes)
  for (int c0 = 0; c0 < VOCAB; c0 += CHUNK) {
    v8f acc[2][4];
#pragma unroll
    for (int mt = 0; mt < 2; ++mt)
#pragma unroll
      for (int nt = 0; nt < 4; ++nt) acc[mt][nt] = v8f{};

#pragma unroll
    for (int kh = 0; kh < 2; ++kh) {
      const uint32_t cur = pb;
      // Kick off the next phase's staging into the other buffer.
      const bool hasNext = !(kh == 1 && c0 + CHUNK >= VOCAB);
      if (hasNext) {
        int nc0 = (kh == 0) ? c0 : c0 + CHUNK;
        stage(nc0, kh ^ 1, cur ^ BUFBYTES);
      }

      // Compute 8 K-steps (K=256) against the current buffer.
      for (int kk = 0; kk < 8; ++kk) {
        const int kloc = kk * 32;               // K within buffer
        const int kbg  = kh * KHALF + kloc;     // K within full row
        // A fragments (contiguous K order, matching B):
        // lanes 0-15 hold K {0..15}; lanes 16-31 hold K {16..31}.
        V16U ah0, al0, ah1, al1;
        {
          const __bf16* p = xh + (size_t)m0 * DD + kbg + half * 16;
          ah0.h[0] = *(const v8bf*)p;
          ah0.h[1] = *(const v8bf*)(p + 8);
          const __bf16* p1 = p + (size_t)16 * DD;
          ah1.h[0] = *(const v8bf*)p1;
          ah1.h[1] = *(const v8bf*)(p1 + 8);
        }
        {
          const __bf16* p = xl + (size_t)m0 * DD + kbg + half * 16;
          al0.h[0] = *(const v8bf*)p;
          al0.h[1] = *(const v8bf*)(p + 8);
          const __bf16* p1 = p + (size_t)16 * DD;
          al1.h[0] = *(const v8bf*)p1;
          al1.h[1] = *(const v8bf*)(p1 + 8);
        }
#pragma unroll
        for (int nt = 0; nt < 4; ++nt) {
          const int col = nwave * 64 + nt * 16 + l15;   // B lane = codebook row
          const char* bbase = smem + cur + (size_t)(col * KHALF + kloc + half * 16) * 2;
          v16bf bh = *(const v16bf*)bbase;
          v16bf bl = *(const v16bf*)(bbase + 65536);
          // 3-term compensated product: Ah*Bh + Ah*Bl + Al*Bh
          acc[0][nt] = wmma_bf16(ah0.v, bh, acc[0][nt]);
          acc[1][nt] = wmma_bf16(ah1.v, bh, acc[1][nt]);
          acc[0][nt] = wmma_bf16(ah0.v, bl, acc[0][nt]);
          acc[1][nt] = wmma_bf16(ah1.v, bl, acc[1][nt]);
          acc[0][nt] = wmma_bf16(al0.v, bh, acc[0][nt]);
          acc[1][nt] = wmma_bf16(al1.v, bh, acc[1][nt]);
        }
      }

      wait_async0();       // my staging writes for next phase have landed
      __syncthreads();     // everyone done reading cur + staging visible
      pb ^= BUFBYTES;
    }

    // ---- fold this chunk into the running argmin: score = ||c||^2 - 2 x.c ----
#pragma unroll
    for (int nt = 0; nt < 4; ++nt) {
      const int colg = c0 + nwave * 64 + nt * 16 + l15;
      const float cc = c2[colg];
#pragma unroll
      for (int mt = 0; mt < 2; ++mt)
#pragma unroll
        for (int r = 0; r < 8; ++r) {
          float s = fmaf(-2.0f, acc[mt][nt][r], cc);
          if (s < best[mt][r]) { best[mt][r] = s; bidx[mt][r] = colg; }
        }
    }
  }

  // ---- butterfly argmin across the 16-lane N stripes (wave32) ----
#pragma unroll
  for (int mt = 0; mt < 2; ++mt)
#pragma unroll
    for (int r = 0; r < 8; ++r) {
      float b = best[mt][r];
      int   i = bidx[mt][r];
#pragma unroll
      for (int off = 1; off < 16; off <<= 1) {
        float ob = __shfl_xor(b, off, 32);
        int   oi = __shfl_xor(i, off, 32);
        if (ob < b || (ob == b && oi < i)) { b = ob; i = oi; }
      }
      if (l15 == 0) {
        // C/D layout: lanes<16 -> M=r, lanes>=16 -> M=r+8
        int mloc = mwave * 32 + mt * 16 + half * 8 + r;
        idxS[mloc]            = i;
        outIdx[wgRow0 + mloc] = (float)i;
        used[i]               = 1u;
      }
    }
  __syncthreads();

  // ---- gather winning codebook rows (fp32) to the embed output ----
  for (int row = wave; row < 128; row += 8) {
    const int idx = idxS[row];
    const float4* src = (const float4*)(cb + (size_t)idx * DD);
    float4* dst = (float4*)(outEmbed + (size_t)(wgRow0 + row) * DD);
#pragma unroll
    for (int j = 0; j < 4; ++j) dst[lane + j * 32] = src[lane + j * 32];
  }
}

// ---------------------------------------------------------------------------
// Tail: new_streak[v] = used[v] ? 0 : streak[v] + 1
// ---------------------------------------------------------------------------
__global__ __launch_bounds__(256) void vq_streak(const int* __restrict__ streak_in,
                                                 const unsigned* __restrict__ used,
                                                 float* __restrict__ outStreak) {
  int v = blockIdx.x * 256 + threadIdx.x;
  if (v < VOCAB) outStreak[v] = used[v] ? 0.0f : (float)(streak_in[v] + 1);
}

// ---------------------------------------------------------------------------
extern "C" void kernel_launch(void* const* d_in, const int* in_sizes, int n_in,
                              void* d_out, int out_size, void* d_ws, size_t ws_size,
                              hipStream_t stream) {
  (void)in_sizes; (void)n_in; (void)out_size; (void)ws_size;
  const float* x      = (const float*)d_in[0];
  const float* cb     = (const float*)d_in[1];
  const int*   streak = (const int*)d_in[2];

  float* out       = (float*)d_out;
  float* outEmbed  = out;                                  // 16384*512
  float* outIdx    = out + (size_t)NROWS * DD;             // 16384
  float* outStreak = outIdx + NROWS;                       // 8192

  // Workspace layout (~48 MB):
  char*     ws   = (char*)d_ws;
  unsigned* used = (unsigned*)ws;                                     // 32 KB
  float*    c2   = (float*)(ws + 32768);                              // 32 KB
  __bf16*   ch   = (__bf16*)(ws + 65536);                             // 8 MB
  __bf16*   cl   = ch + (size_t)VOCAB * DD;                           // 8 MB
  __bf16*   xh   = (__bf16*)(ws + 65536 + (size_t)4 * VOCAB * DD);    // 16 MB
  __bf16*   xl   = xh + (size_t)NROWS * DD;                           // 16 MB

  const int lds_bytes = 2 * BUFBYTES + 128 * (int)sizeof(int);  // 256KB + idx
  (void)hipFuncSetAttribute((const void*)vq_main,
                            hipFuncAttributeMaxDynamicSharedMemorySize, lds_bytes);

  vq_prep_x <<<(NROWS * DD / 4) / 256, 256, 0, stream>>>(x, xh, xl);
  vq_prep_cb<<<VOCAB / 8,              256, 0, stream>>>(cb, ch, cl, c2, used);
  vq_main   <<<NROWS / 128, 256, lds_bytes, stream>>>(xh, xl, ch, cl, c2, cb,
                                                      outEmbed, outIdx, used);
  vq_streak <<<VOCAB / 256,            256, 0, stream>>>(streak, used, outStreak);
}
